// GNN_SR_Net_61881888800771
// MI455X (gfx1250) — matrix operand order repaired and checked by reading, not compiled, not run on hardware
//
#include <hip/hip_runtime.h>
#include <math.h>

#define N_NODES   100000
#define NEDGES    600000
#define NSEDGES   150000
#define DIM       128
#define NREL      4
#define DMODEL    384
#define HEADS     4
#define DH        96
#define BZ        1024
#define SEQ       50
#define NPRED     6
#define KCAT      640           // 128 (root) + 4*128 (relations)
#define NEGV      (-4294967295.0f)

typedef __bf16        v16bf __attribute__((ext_vector_type(16)));
typedef float         v8f   __attribute__((ext_vector_type(8)));
typedef float         v4f   __attribute__((ext_vector_type(4)));
typedef unsigned int  v4u   __attribute__((ext_vector_type(4)));

#define WMMA_BF16(a, b, c) \
  __builtin_amdgcn_wmma_f32_16x16x32_bf16(false, (a), false, (b), (short)0, (c), false, false)

// ---------------- WMMA fragment loaders (CDNA5 16x16x32 bf16 layout) -------
// A 16x32: lane L holds row M=L&15; elems 0..7 -> K = k0 + (L>>4)*8 + 0..7,
//          elems 8..15 -> K = k0 + 16 + (L>>4)*8 + 0..7.
__device__ inline v16bf cvt_a_frag(const float* p) {
  v4f f0 = ((const v4f*)p)[0];
  v4f f1 = ((const v4f*)p)[1];
  v4f f2 = ((const v4f*)(p + 16))[0];
  v4f f3 = ((const v4f*)(p + 16))[1];
  v16bf a;
#pragma unroll
  for (int i = 0; i < 4; ++i) {
    a[i]      = (__bf16)f0[i];
    a[4 + i]  = (__bf16)f1[i];
    a[8 + i]  = (__bf16)f2[i];
    a[12 + i] = (__bf16)f3[i];
  }
  return a;
}
// B 32x16 from pre-transposed bf16 Bt (N x K row-major): lane L holds column
// n=L&15, elems 0..7 -> K = k0 + (L>>4)*8 + e, elems 8..15 -> K = k0 + 16 + (L>>4)*8 + (e-8)
__device__ inline v16bf load_b_frag(const __bf16* p) {
  v16bf r;
  ((v4u*)&r)[0] = ((const v4u*)p)[0];
  ((v4u*)&r)[1] = ((const v4u*)(p + 16))[0];
  return r;
}

// ---------------- generic bf16 WMMA GEMM: C[M,N] = A[M,K] @ Bt^T ----------
// 32x64 macro-tile per wave: 2 A-frags x 4 B-frags -> 8 WMMAs per k-step.
// mode 0: C=acc(+bias)   mode 1: relu(acc+bias)   mode 2: acc+bias+addsrc
__global__ void gemm_bf16(const float* __restrict__ A, int lda,
                          const __bf16* __restrict__ Bt,
                          float* __restrict__ C,
                          const float* __restrict__ bias,
                          const float* __restrict__ addsrc,
                          int M, int N, int K, int mode) {
  int wave = threadIdx.x >> 5;
  int lane = threadIdx.x & 31;
  int m0 = (blockIdx.x * 8 + wave) * 32;
  int n0 = blockIdx.y * 64;
  if (m0 >= M) return;
  int half = lane >> 4;
  int lc = lane & 15;
  const float* arow0 = A + (size_t)(m0 + lc) * lda + half * 8;
  const float* arow1 = arow0 + (size_t)16 * lda;
  const __bf16* brow = Bt + (size_t)(n0 + lc) * K + half * 8;
  const size_t bstep = (size_t)16 * K;
  v8f acc[2][4] = {};
  for (int k0 = 0; k0 < K; k0 += 32) {
    v16bf a0 = cvt_a_frag(arow0 + k0);
    v16bf a1 = cvt_a_frag(arow1 + k0);
    v16bf b0 = load_b_frag(brow + k0);
    v16bf b1 = load_b_frag(brow + bstep + k0);
    v16bf b2 = load_b_frag(brow + 2 * bstep + k0);
    v16bf b3 = load_b_frag(brow + 3 * bstep + k0);
    acc[0][0] = WMMA_BF16(a0, b0, acc[0][0]);
    acc[0][1] = WMMA_BF16(a0, b1, acc[0][1]);
    acc[0][2] = WMMA_BF16(a0, b2, acc[0][2]);
    acc[0][3] = WMMA_BF16(a0, b3, acc[0][3]);
    acc[1][0] = WMMA_BF16(a1, b0, acc[1][0]);
    acc[1][1] = WMMA_BF16(a1, b1, acc[1][1]);
    acc[1][2] = WMMA_BF16(a1, b2, acc[1][2]);
    acc[1][3] = WMMA_BF16(a1, b3, acc[1][3]);
  }
#pragma unroll
  for (int mi = 0; mi < 2; ++mi) {
    int rb = m0 + mi * 16 + half * 8;
#pragma unroll
    for (int j = 0; j < 4; ++j) {
      int col = n0 + j * 16 + lc;
      float bi = bias ? bias[col] : 0.0f;
#pragma unroll
      for (int v = 0; v < 8; ++v) {
        size_t idx = (size_t)(rb + v) * N + col;
        float val = acc[mi][j][v] + bi;
        if (mode == 1)      val = fmaxf(val, 0.0f);
        else if (mode == 2) val += addsrc[idx];
        C[idx] = val;
      }
    }
  }
}

// ---------------- RGCN --------------------------------------------------
// Wcat_bt: 128 x 640 bf16; k<128 -> root[k][n]; else W_r[i][n]=sum_b c*basis
__global__ void build_wcat(const float* __restrict__ basis,
                           const float* __restrict__ coeff,
                           const float* __restrict__ root,
                           __bf16* __restrict__ Bt) {
  int idx = blockIdx.x * blockDim.x + threadIdx.x;
  if (idx >= DIM * KCAT) return;
  int k = idx % KCAT, n = idx / KCAT;
  float v;
  if (k < DIM) {
    v = root[k * DIM + n];
  } else {
    int r = (k - DIM) >> 7, i = k & 127;
    v = coeff[r * 2 + 0] * basis[0 * DIM * DIM + i * DIM + n] +
        coeff[r * 2 + 1] * basis[1 * DIM * DIM + i * DIM + n];
  }
  Bt[idx] = (__bf16)v;
}

// per-relation scatter-add: S[r][dst] += x[src]; one wave per edge, 4 f per lane
__global__ void rgcn_scatter(const float* __restrict__ x, int ldx,
                             const int* __restrict__ ei,
                             const int* __restrict__ et,
                             int nE, float* __restrict__ S) {
  int wave  = (blockIdx.x * blockDim.x + threadIdx.x) >> 5;
  int lane  = threadIdx.x & 31;
  int waves = (gridDim.x * blockDim.x) >> 5;
  for (int e = wave; e < nE; e += waves) {
    int src = ei[e], dst = ei[nE + e], r = et[e];
    v4f v = ((const v4f*)(x + (size_t)src * ldx))[lane];
    float* sd = S + ((size_t)r * N_NODES + dst) * DIM + lane * 4;
    atomicAdd(sd + 0, v[0]);
    atomicAdd(sd + 1, v[1]);
    atomicAdd(sd + 2, v[2]);
    atomicAdd(sd + 3, v[3]);
  }
}

// virtual A row: k<128 -> x[m,k]; else S[r][m][k&127]; 8+8 group stays in block
__device__ inline const float* rgcn_ptr(const float* x, int ldx,
                                        const float* S, int m, int k) {
  if (k < DIM) return x + (size_t)m * ldx + k;
  int r = (k - DIM) >> 7;
  return S + ((size_t)r * N_NODES + m) * DIM + (k & 127);
}

// fused [x | S0..S3] @ [root; W0..W3] with tanh epilogue; 32x64 tile per wave
__global__ void rgcn_gemm(const float* __restrict__ x, int ldx,
                          const float* __restrict__ S,
                          const __bf16* __restrict__ Wbt,
                          float* __restrict__ out /* cs + colofs, ld=384 */) {
  int wave = threadIdx.x >> 5;
  int lane = threadIdx.x & 31;
  int m0 = (blockIdx.x * 8 + wave) * 32;
  int n0 = blockIdx.y * 64;
  if (m0 >= N_NODES) return;
  int half = lane >> 4;
  int lc = lane & 15;
  int ma = m0 + lc;
  int mb = ma + 16;
  const __bf16* brow = Wbt + (size_t)(n0 + lc) * KCAT + half * 8;
  const size_t bstep = (size_t)16 * KCAT;
  v8f acc[2][4] = {};
  for (int k0 = 0; k0 < KCAT; k0 += 32) {
    int ksub = k0 + half * 8;
    v16bf a0 = cvt_a_frag(rgcn_ptr(x, ldx, S, ma, ksub));
    v16bf a1 = cvt_a_frag(rgcn_ptr(x, ldx, S, mb, ksub));
    v16bf b0 = load_b_frag(brow + k0);
    v16bf b1 = load_b_frag(brow + bstep + k0);
    v16bf b2 = load_b_frag(brow + 2 * bstep + k0);
    v16bf b3 = load_b_frag(brow + 3 * bstep + k0);
    acc[0][0] = WMMA_BF16(a0, b0, acc[0][0]);
    acc[0][1] = WMMA_BF16(a0, b1, acc[0][1]);
    acc[0][2] = WMMA_BF16(a0, b2, acc[0][2]);
    acc[0][3] = WMMA_BF16(a0, b3, acc[0][3]);
    acc[1][0] = WMMA_BF16(a1, b0, acc[1][0]);
    acc[1][1] = WMMA_BF16(a1, b1, acc[1][1]);
    acc[1][2] = WMMA_BF16(a1, b2, acc[1][2]);
    acc[1][3] = WMMA_BF16(a1, b3, acc[1][3]);
  }
#pragma unroll
  for (int mi = 0; mi < 2; ++mi) {
    int rb = m0 + mi * 16 + half * 8;
#pragma unroll
    for (int j = 0; j < 4; ++j) {
      int col = n0 + j * 16 + lc;
#pragma unroll
      for (int v = 0; v < 8; ++v)
        out[(size_t)(rb + v) * DMODEL + col] = tanhf(acc[mi][j][v]);
    }
  }
}

// ---------------- misc element kernels ----------------------------------
__global__ void w_to_bt(const float* __restrict__ W, __bf16* __restrict__ Bt,
                        int K, int N) {
  int idx = blockIdx.x * blockDim.x + threadIdx.x;
  if (idx >= K * N) return;
  int k = idx % K, n = idx / K;
  Bt[(size_t)n * K + k] = (__bf16)W[(size_t)k * N + n];
}

__global__ void gather_user(const float* __restrict__ cs,
                            const int* __restrict__ users,
                            float* __restrict__ ue) {
  int b = blockIdx.x, d = threadIdx.x;
  ue[(size_t)b * DMODEL + d] = cs[(size_t)users[b] * DMODEL + d];
}

__global__ void gather_seq_pe(const float* __restrict__ cs,
                              const int* __restrict__ seqs,
                              float* __restrict__ inp) {
  int bs = blockIdx.x, d = threadIdx.x;
  int s = bs % SEQ;
  int node = seqs[bs];
  float div = expf((float)(d & ~1) * (-9.210340371976184f / (float)DMODEL));
  float ang = (float)s * div;
  float pe = (d & 1) ? cosf(ang) : sinf(ang);
  inp[(size_t)bs * DMODEL + d] = cs[(size_t)node * DMODEL + d] + pe;
}

// scores + softmax over q (axis=2 of (h,b,q,k)) for one (b,h,k); a[(b,h,k),q]
__global__ void attn_scores_softmax(const float* __restrict__ Qp,
                                    const float* __restrict__ Kp,
                                    float* __restrict__ a) {
  int bhk = blockIdx.x;
  int k = bhk % SEQ, bh = bhk / SEQ;
  int h = bh % HEADS, b = bh / HEADS;
  int q = threadIdx.x;
  __shared__ float sv[64];
  const float* kv = Kp + ((size_t)b * SEQ + k) * DMODEL + h * DH;
  float s = NEGV;
  if (q < SEQ && q >= k) {
    const float* qv = Qp + ((size_t)b * SEQ + q) * DMODEL + h * DH;
    float dot = 0.0f;
#pragma unroll 8
    for (int d = 0; d < DH; ++d) dot += qv[d] * kv[d];
    s = dot * 0.14142135623730951f;   // 1/sqrt(50)
  }
  sv[threadIdx.x] = s;
  __syncthreads();
  for (int st = 32; st > 0; st >>= 1) {
    if (threadIdx.x < st) sv[threadIdx.x] = fmaxf(sv[threadIdx.x], sv[threadIdx.x + st]);
    __syncthreads();
  }
  float mx = sv[0];
  __syncthreads();
  float e = (q < SEQ) ? expf(s - mx) : 0.0f;   // masked -> exp(NEG-mx)=0
  sv[threadIdx.x] = e;
  __syncthreads();
  for (int st = 32; st > 0; st >>= 1) {
    if (threadIdx.x < st) sv[threadIdx.x] += sv[threadIdx.x + st];
    __syncthreads();
  }
  if (q < SEQ)
    a[((size_t)(b * HEADS + h) * SEQ + k) * SEQ + q] = e / sv[0];
}

// att[b,q,h*96+d] = sum_k a[(b,h,k),q] * V[b,k,h*96+d]
__global__ void attn_apply(const float* __restrict__ a,
                           const float* __restrict__ Vp,
                           float* __restrict__ att) {
  int d = threadIdx.x;
  int bq = blockIdx.x;
  int q = bq % SEQ, b = bq / SEQ;
  int h = d / DH;
  const float* ab = a + ((size_t)(b * HEADS + h) * SEQ) * SEQ + q;
  float acc = 0.0f;
  for (int k = 0; k < SEQ; ++k)
    acc += ab[(size_t)k * SEQ] * Vp[((size_t)b * SEQ + k) * DMODEL + d];
  att[(size_t)b * SEQ * DMODEL + (size_t)q * DMODEL + d] = acc;
}

// out[row] = LN(Y[row]) * g + bl + addv[row]     (row length = 384)
__global__ void ln_add(const float* __restrict__ Y,
                       const float* __restrict__ g,
                       const float* __restrict__ bl,
                       const float* __restrict__ addv,
                       float* __restrict__ out) {
  int row = blockIdx.x, d = threadIdx.x;
  __shared__ float sh[DMODEL];
  float y = Y[(size_t)row * DMODEL + d];
  sh[d] = y;
  __syncthreads();
  if (d < 128) sh[d] += sh[d + 256];
  __syncthreads();
  for (int st = 128; st > 0; st >>= 1) {
    if (d < st) sh[d] += sh[d + st];
    __syncthreads();
  }
  float mu = sh[0] * (1.0f / DMODEL);
  __syncthreads();
  float c = y - mu;
  sh[d] = c * c;
  __syncthreads();
  if (d < 128) sh[d] += sh[d + 256];
  __syncthreads();
  for (int st = 128; st > 0; st >>= 1) {
    if (d < st) sh[d] += sh[d + st];
    __syncthreads();
  }
  float var = sh[0] * (1.0f / DMODEL);
  float r = rsqrtf(var + 1e-8f);
  out[(size_t)row * DMODEL + d] =
      c * r * g[d] + bl[d] + addv[(size_t)row * DMODEL + d];
}

__global__ void seq_sum(const float* __restrict__ ie, float* __restrict__ se) {
  int b = blockIdx.x, d = threadIdx.x;
  float acc = 0.0f;
  for (int k = 0; k < SEQ; ++k)
    acc += ie[((size_t)b * SEQ + k) * DMODEL + d];
  se[(size_t)b * DMODEL + d] = acc;
}

__global__ void predict(const float* __restrict__ ue,
                        const float* __restrict__ ie2,
                        const float* __restrict__ pred_w,
                        const float* __restrict__ pred_b,
                        const int* __restrict__ items,
                        float* __restrict__ out) {
  int i = blockIdx.x * blockDim.x + threadIdx.x;
  if (i >= BZ * NPRED) return;
  int b = i / NPRED;
  int item = items[i];
  const float* pw = pred_w + (size_t)item * DMODEL;
  const float* u  = ue  + (size_t)b * DMODEL;
  const float* e2 = ie2 + (size_t)b * DMODEL;
  float acc = pred_b[item];
#pragma unroll 8
  for (int d = 0; d < DMODEL; ++d) acc += pw[d] * (u[d] + e2[d]);
  out[i] = acc;
}

// ======================= host side =======================================
extern "C" void kernel_launch(void* const* d_in, const int* in_sizes, int n_in,
                              void* d_out, int out_size, void* d_ws, size_t ws_size,
                              hipStream_t stream) {
  (void)in_sizes; (void)n_in; (void)out_size; (void)ws_size;
  const float* node_emb   = (const float*)d_in[0];
  const float* conv_bases = (const float*)d_in[1];
  const float* conv_coeffs= (const float*)d_in[2];
  const float* conv_roots = (const float*)d_in[3];
  const float* s_basis    = (const float*)d_in[4];
  const float* s_coeff    = (const float*)d_in[5];
  const float* s_root     = (const float*)d_in[6];
  const float* tsal_wq    = (const float*)d_in[7];
  const float* tsal_wk    = (const float*)d_in[8];
  const float* tsal_wv    = (const float*)d_in[9];
  const float* ff1_w1     = (const float*)d_in[10];
  const float* ff1_b1     = (const float*)d_in[11];
  const float* ff1_w2     = (const float*)d_in[12];
  const float* ff1_b2     = (const float*)d_in[13];
  const float* ln1_g      = (const float*)d_in[14];
  const float* ln1_b      = (const float*)d_in[15];
  // cal_wq (16), cal_wk (17) are dead: softmax over size-1 axis => weights==1
  const float* cal_wv     = (const float*)d_in[18];
  const float* ff2_w1     = (const float*)d_in[19];
  const float* ff2_b1     = (const float*)d_in[20];
  const float* ff2_w2     = (const float*)d_in[21];
  const float* ff2_b2     = (const float*)d_in[22];
  const float* ln2_g      = (const float*)d_in[23];
  const float* ln2_b      = (const float*)d_in[24];
  const float* pred_w     = (const float*)d_in[25];
  const float* pred_b     = (const float*)d_in[26];
  const int* edge_index   = (const int*)d_in[27];
  const int* edge_type    = (const int*)d_in[28];
  const int* sedge_index  = (const int*)d_in[29];
  const int* sedge_type   = (const int*)d_in[30];
  const int* batch_users  = (const int*)d_in[31];
  const int* batch_seqs   = (const int*)d_in[32];
  const int* items_pred   = (const int*)d_in[33];

  // ---- workspace carving (floats) ----
  float* fb = (float*)d_ws;
  size_t off = 0;
  float* cs   = fb + off; off += (size_t)N_NODES * DMODEL;       // 38.4M
  float* S    = fb + off; off += (size_t)NREL * N_NODES * DIM;   // 51.2M
  float* inp  = fb + off; off += (size_t)BZ * SEQ * DMODEL;
  float* Vp   = fb + off; off += (size_t)BZ * SEQ * DMODEL;      // -> item_embs
  float* att  = fb + off; off += (size_t)BZ * SEQ * DMODEL;      // -> Y1 in place
  float* ue   = fb + off; off += (size_t)BZ * DMODEL;
  float* se   = fb + off; off += (size_t)BZ * DMODEL;
  float* catt = fb + off; off += (size_t)BZ * DMODEL;            // -> y2 in place
  float* h2   = fb + off; off += (size_t)BZ * DMODEL;
  float* ie2  = fb + off; off += (size_t)BZ * DMODEL;
  __bf16* bt = (__bf16*)(fb + off);
  __bf16* wq_bt  = bt; bt += DMODEL * DMODEL;
  __bf16* wk_bt  = bt; bt += DMODEL * DMODEL;
  __bf16* wv_bt  = bt; bt += DMODEL * DMODEL;
  __bf16* w11_bt = bt; bt += DMODEL * DMODEL;
  __bf16* w12_bt = bt; bt += DMODEL * DMODEL;
  __bf16* wvc_bt = bt; bt += DMODEL * DMODEL;
  __bf16* w21_bt = bt; bt += DMODEL * DMODEL;
  __bf16* w22_bt = bt; bt += DMODEL * DMODEL;
  __bf16* wcat_bt = bt; bt += DIM * KCAT;
  // reuse dead regions
  float* Qp     = S;                                   // after RGCN, S is dead
  float* Kp     = S + (size_t)BZ * SEQ * DMODEL;
  float* ascore = S + 2 * (size_t)BZ * SEQ * DMODEL;   // 10.24M fits
  float* H1     = cs;                                  // cs dead after gathers

  auto gemm = [&](const float* A, int lda, const __bf16* Bt, float* C,
                  const float* bias, const float* add, int M, int N, int K,
                  int mode) {
    dim3 g((M + 255) / 256, N / 64);
    gemm_bf16<<<g, 256, 0, stream>>>(A, lda, Bt, C, bias, add, M, N, K, mode);
  };
  auto cw = [&](const float* W, __bf16* Bt, int K, int N) {
    w_to_bt<<<(K * N + 255) / 256, 256, 0, stream>>>(W, Bt, K, N);
  };

  // ---- weight conversions (bf16, transposed for B-fragments) ----
  cw(tsal_wq, wq_bt, DMODEL, DMODEL);
  cw(tsal_wk, wk_bt, DMODEL, DMODEL);
  cw(tsal_wv, wv_bt, DMODEL, DMODEL);
  cw(ff1_w1, w11_bt, DMODEL, DMODEL);
  cw(ff1_w2, w12_bt, DMODEL, DMODEL);
  cw(cal_wv, wvc_bt, DMODEL, DMODEL);
  cw(ff2_w1, w21_bt, DMODEL, DMODEL);
  cw(ff2_w2, w22_bt, DMODEL, DMODEL);

  // ---- RGCN layers (scatter-first + fused K=640 WMMA GEMM, tanh) ----
  dim3 gr((N_NODES + 255) / 256, DIM / 64);
  const size_t sbytes = (size_t)NREL * N_NODES * DIM * sizeof(float);
  // layer 1: in = node_emb (ld 128) -> cs[:,0:128]
  build_wcat<<<(DIM * KCAT + 255) / 256, 256, 0, stream>>>(
      conv_bases, conv_coeffs, conv_roots, wcat_bt);
  hipMemsetAsync(S, 0, sbytes, stream);
  rgcn_scatter<<<4096, 256, 0, stream>>>(node_emb, DIM, edge_index, edge_type,
                                         NEDGES, S);
  rgcn_gemm<<<gr, 256, 0, stream>>>(node_emb, DIM, S, wcat_bt, cs + 0);
  // layer 2: in = cs[:,0:128] (ld 384) -> cs[:,128:256]
  build_wcat<<<(DIM * KCAT + 255) / 256, 256, 0, stream>>>(
      conv_bases + 2 * DIM * DIM, conv_coeffs + NREL * 2,
      conv_roots + DIM * DIM, wcat_bt);
  hipMemsetAsync(S, 0, sbytes, stream);
  rgcn_scatter<<<4096, 256, 0, stream>>>(cs, DMODEL, edge_index, edge_type,
                                         NEDGES, S);
  rgcn_gemm<<<gr, 256, 0, stream>>>(cs, DMODEL, S, wcat_bt, cs + DIM);
  // layer 3 (short graph): in = cs[:,128:256] -> cs[:,256:384]
  build_wcat<<<(DIM * KCAT + 255) / 256, 256, 0, stream>>>(s_basis, s_coeff,
                                                           s_root, wcat_bt);
  hipMemsetAsync(S, 0, sbytes, stream);
  rgcn_scatter<<<4096, 256, 0, stream>>>(cs + DIM, DMODEL, sedge_index,
                                         sedge_type, NSEDGES, S);
  rgcn_gemm<<<gr, 256, 0, stream>>>(cs + DIM, DMODEL, S, wcat_bt, cs + 2 * DIM);

  // ---- gathers (+ sinusoidal PE) ----
  gather_user<<<BZ, DMODEL, 0, stream>>>(cs, batch_users, ue);
  gather_seq_pe<<<BZ * SEQ, DMODEL, 0, stream>>>(cs, batch_seqs, inp);

  // ---- self-attention ----
  const int MT = BZ * SEQ;  // 51200 token rows
  gemm(inp, DMODEL, wq_bt, Qp, nullptr, nullptr, MT, DMODEL, DMODEL, 0);
  gemm(inp, DMODEL, wk_bt, Kp, nullptr, nullptr, MT, DMODEL, DMODEL, 0);
  gemm(inp, DMODEL, wv_bt, Vp, nullptr, nullptr, MT, DMODEL, DMODEL, 0);
  attn_scores_softmax<<<BZ * HEADS * SEQ, 64, 0, stream>>>(Qp, Kp, ascore);
  attn_apply<<<BZ * SEQ, DMODEL, 0, stream>>>(ascore, Vp, att);

  // ---- FFN1 + LN + residuals: item_embs = LN(att + FFN(att)) + inp ----
  gemm(att, DMODEL, w11_bt, H1, ff1_b1, nullptr, MT, DMODEL, DMODEL, 1);
  gemm(H1, DMODEL, w12_bt, att, ff1_b2, att, MT, DMODEL, DMODEL, 2);  // Y1 in place
  ln_add<<<MT, DMODEL, 0, stream>>>(att, ln1_g, ln1_b, inp, Vp);  // item_embs -> Vp

  // ---- cross-attention collapses to column-sum @ cal_wv ----
  seq_sum<<<BZ, DMODEL, 0, stream>>>(Vp, se);
  gemm(se, DMODEL, wvc_bt, catt, nullptr, nullptr, BZ, DMODEL, DMODEL, 0);
  gemm(catt, DMODEL, w21_bt, h2, ff2_b1, nullptr, BZ, DMODEL, DMODEL, 1);
  gemm(h2, DMODEL, w22_bt, catt, ff2_b2, catt, BZ, DMODEL, DMODEL, 2); // y2 in place
  ln_add<<<BZ, DMODEL, 0, stream>>>(catt, ln2_g, ln2_b, ue, ie2);

  // ---- prediction: res[b,t] = pw·(ue+ie2) + pb ----
  predict<<<(BZ * NPRED + 255) / 256, 256, 0, stream>>>(
      ue, ie2, pred_w, pred_b, items_pred, (float*)d_out);
}
